// DeltaUnit_37065567764610
// MI455X (gfx1250) — compile-verified, matrix operand order checked by reading, not run on hardware
//
#include <hip/hip_runtime.h>
#include <hip/hip_bf16.h>
#include <cstdint>
#include <cstddef>

// ---------------------------------------------------------------------------
// Types for CDNA5 (gfx1250) wave32 WMMA
// ---------------------------------------------------------------------------
typedef __bf16 bf16_t;
typedef __attribute__((ext_vector_type(16))) __bf16 bf16x16;
typedef __attribute__((ext_vector_type(8)))  __bf16 bf16x8;
typedef __attribute__((ext_vector_type(8)))  float  f32x8;

#define DEV __device__ __forceinline__

// ---------------------------------------------------------------------------
// CDNA5 async memory->LDS DMA (ASYNCcnt-tracked, no VGPR data path).
// SADDR form: global_load_async_to_lds_b128 vdst_lds, voffset, s[base:base+1]
// LDS byte address = low 32 bits of the generic pointer (ISA 10.2 aperture rule).
// ---------------------------------------------------------------------------
DEV void async_b128(const void* sbase, uint32_t voff, void* lds) {
  asm volatile("global_load_async_to_lds_b128 %0, %1, %2"
               :
               : "v"((uint32_t)(uintptr_t)lds), "v"(voff),
                 "s"((uint64_t)(uintptr_t)sbase)
               : "memory");
}
DEV void wait_async0() { asm volatile("s_wait_asynccnt 0x0" ::: "memory"); }
DEV void wait_async4() { asm volatile("s_wait_asynccnt 0x4" ::: "memory"); }

// A/B fragment loader (16-bit, 16x32 layout, wave32):
//   lane L: row = row0 + (L & 15); K-half = (L >> 4)
//   elems 0..7  -> K = 8*half + 0..7          (contiguous 16B)
//   elems 8..15 -> K = 16 + 8*half + 0..7     (contiguous 16B at +16 elems)
DEV bf16x16 frag_ld(const bf16_t* base, int ld, int row0) {
  const int lane = threadIdx.x & 31;
  const int r    = row0 + (lane & 15);
  const int kh   = (lane >> 4) << 3;
  const bf16_t* p = base + r * ld + kh;
  bf16x8 lo = *(const bf16x8*)(p);
  bf16x8 hi = *(const bf16x8*)(p + 16);
  return __builtin_shufflevector(lo, hi, 0,1,2,3,4,5,6,7,8,9,10,11,12,13,14,15);
}

DEV f32x8 wmma_bf16(bf16x16 a, bf16x16 b, f32x8 c) {
  return __builtin_amdgcn_wmma_f32_16x16x32_bf16(
      /*neg_a=*/false, a, /*neg_b=*/false, b,
      /*c_mod=*/(short)0, c, /*reuse_a=*/false, /*reuse_b=*/false);
}

// C/D fragment (16x16 f32): lane L -> col = col0 + (L&15); VGPR r -> row = r + 8*(L>>4)
DEV f32x8 cfrag_ld(const float* base, int ld, int row0, int col0) {
  const int lane = threadIdx.x & 31;
  const int col  = col0 + (lane & 15);
  const int rb   = row0 + ((lane >> 4) << 3);
  f32x8 c;
#pragma unroll
  for (int r = 0; r < 8; ++r) c[r] = base[(size_t)(rb + r) * ld + col];
  return c;
}

DEV void cfrag_st(float* base, int ld, int row0, int col0, f32x8 c) {
  const int lane = threadIdx.x & 31;
  const int col  = col0 + (lane & 15);
  const int rb   = row0 + ((lane >> 4) << 3);
#pragma unroll
  for (int r = 0; r < 8; ++r) base[(size_t)(rb + r) * ld + col] = c[r];
}

// ---------------------------------------------------------------------------
// fp32 -> bf16 conversion
// ---------------------------------------------------------------------------
__global__ void __launch_bounds__(256) f2bf_kernel(const float* __restrict__ in,
                                                   bf16_t* __restrict__ out, size_t n) {
  size_t i = (size_t)blockIdx.x * 256 + threadIdx.x;
  size_t stride = (size_t)gridDim.x * 256;
  for (; i < n; i += stride) out[i] = (bf16_t)in[i];
}

// ---------------------------------------------------------------------------
// Generic WMMA GEMM:  Y[M,N] = act( A[M,K] * W[N,K]^T )
// 128x128 block tile, BK=32 (== one WMMA K step), 8 waves: 4(M) x 2(N),
// each wave computes a 32x64 sub-tile = 2x4 fragments.
// Double-buffered LDS staging via global_load_async_to_lds_b128:
//   issue stage(kt+1) -> s_wait_asynccnt 4 (stage kt retired, in-order) -> WMMA
// ACT: 0=none 1=silu 2=sigmoid 3=2*sigmoid ; OUTBF: 1 -> bf16 out, 0 -> f32
// ---------------------------------------------------------------------------
template<int ACT, int OUTBF>
__global__ void __launch_bounds__(256)
gemm_wmma(const bf16_t* __restrict__ A, const bf16_t* __restrict__ W,
          float* __restrict__ Cf, bf16_t* __restrict__ Cb,
          int M, int N, int K) {
  __shared__ bf16_t As[2][128 * 40];   // 128 rows, 32 K + pad(8) -> 80B row stride
  __shared__ bf16_t Bs[2][128 * 40];

  const int tid  = threadIdx.x;
  const int wave = tid >> 5;
  const int wm   = wave >> 1;       // 0..3
  const int wn   = wave & 1;        // 0..1
  const int m0   = blockIdx.y * 128;
  const int n0   = blockIdx.x * 128;

  // per-thread staging coordinates: 512 16B chunks / 256 threads = 2 each
  const int row0s = (tid + 0)   >> 2, cc0 = ((tid + 0)   & 3) << 3;
  const int row1s = (tid + 256) >> 2, cc1 = ((tid + 256) & 3) << 3;

  f32x8 acc[2][4];
#pragma unroll
  for (int i = 0; i < 2; ++i)
#pragma unroll
    for (int j = 0; j < 4; ++j) acc[i][j] = f32x8{0,0,0,0,0,0,0,0};

  const int nk = K >> 5;

  // 4 async ops per thread per stage (2 for A, 2 for B) == ASYNCcnt += 4 per wave
  auto stage = [&](int bufi, int k0) {
    async_b128(A, (uint32_t)(((size_t)(m0 + row0s) * K + k0 + cc0) * 2),
               &As[bufi][row0s * 40 + cc0]);
    async_b128(W, (uint32_t)(((size_t)(n0 + row0s) * K + k0 + cc0) * 2),
               &Bs[bufi][row0s * 40 + cc0]);
    async_b128(A, (uint32_t)(((size_t)(m0 + row1s) * K + k0 + cc1) * 2),
               &As[bufi][row1s * 40 + cc1]);
    async_b128(W, (uint32_t)(((size_t)(n0 + row1s) * K + k0 + cc1) * 2),
               &Bs[bufi][row1s * 40 + cc1]);
  };

  stage(0, 0);
  for (int kt = 0; kt < nk; ++kt) {
    const int cur = kt & 1;
    if (kt + 1 < nk) {
      stage(cur ^ 1, (kt + 1) << 5);
      if (kt + 2 < nk) {  // CDNA5 global_prefetch_b8 two tiles ahead
        __builtin_prefetch(A + (size_t)(m0 + (tid >> 1)) * K + ((kt + 2) << 5), 0, 1);
        __builtin_prefetch(W + (size_t)(n0 + (tid >> 1)) * K + ((kt + 2) << 5), 0, 1);
      }
      wait_async4();   // stage kt fully retired (async loads complete in order)
    } else {
      wait_async0();
    }
    __syncthreads();

    bf16x16 af[2], bfv[4];
#pragma unroll
    for (int mt = 0; mt < 2; ++mt) af[mt]  = frag_ld(As[cur], 40, wm * 32 + mt * 16);
#pragma unroll
    for (int nt = 0; nt < 4; ++nt) bfv[nt] = frag_ld(Bs[cur], 40, wn * 64 + nt * 16);
#pragma unroll
    for (int mt = 0; mt < 2; ++mt)
#pragma unroll
      for (int nt = 0; nt < 4; ++nt)
        acc[mt][nt] = wmma_bf16(af[mt], bfv[nt], acc[mt][nt]);
    __syncthreads();   // done reading buf[cur] before stage kt+2 overwrites it
  }

  const int lane   = tid & 31;
  const int colb   = lane & 15;
  const int rowoff = (lane >> 4) << 3;
#pragma unroll
  for (int mt = 0; mt < 2; ++mt)
#pragma unroll
    for (int nt = 0; nt < 4; ++nt) {
      int gcol = n0 + wn * 64 + nt * 16 + colb;
#pragma unroll
      for (int r = 0; r < 8; ++r) {
        int grow = m0 + wm * 32 + mt * 16 + rowoff + r;
        float val = acc[mt][nt][r];
        if (ACT == 1)      val = val / (1.f + __expf(-val));
        else if (ACT == 2) val = 1.f / (1.f + __expf(-val));
        else if (ACT == 3) val = 2.f / (1.f + __expf(-val));
        if constexpr (OUTBF) Cb[(size_t)grow * N + gcol] = (bf16_t)val;
        else                 Cf[(size_t)grow * N + gcol] = val;
      }
    }
}

// ---------------------------------------------------------------------------
// beta = 2*sigmoid(x @ Wbeta^T)   [M,H]  (tiny GEMM, VALU)
// ---------------------------------------------------------------------------
__global__ void __launch_bounds__(256)
beta_kernel(const float* __restrict__ x, const float* __restrict__ Wb,
            float* __restrict__ beta, int M, int D, int H) {
  int idx = blockIdx.x * 256 + threadIdx.x;
  if (idx >= M * H) return;
  int m = idx / H, h = idx % H;
  const float* xr = x + (size_t)m * D;
  const float* wr = Wb + (size_t)h * D;
  float s = 0.f;
  for (int i = 0; i < D; ++i) s += xr[i] * wr[i];
  beta[idx] = 2.f / (1.f + __expf(-s));
}

// ---------------------------------------------------------------------------
// L2-normalize each head vector of k (post-silu, fp32) -> bf16
// ---------------------------------------------------------------------------
__global__ void __launch_bounds__(256)
knorm_kernel(const float* __restrict__ kf, bf16_t* __restrict__ kb,
             int nvec, int HD) {
  int idx = blockIdx.x * 256 + threadIdx.x;
  if (idx >= nvec) return;
  const float* p = kf + (size_t)idx * HD;
  float ss = 0.f;
  for (int i = 0; i < HD; ++i) ss += p[i] * p[i];
  float r = rsqrtf(ss + 1e-6f);
  bf16_t* q = kb + (size_t)idx * HD;
  for (int i = 0; i < HD; ++i) q[i] = (bf16_t)(p[i] * r);
}

// ---------------------------------------------------------------------------
// og = RMSNorm_head(o * gate) * g_norm  -> bf16 (ready for final GEMM)
// ---------------------------------------------------------------------------
__global__ void __launch_bounds__(256)
og_kernel(const float* __restrict__ o, const float* __restrict__ gate,
          const float* __restrict__ gnorm, bf16_t* __restrict__ ogb,
          int nvec, int HD, int H) {
  int idx = blockIdx.x * 256 + threadIdx.x;
  if (idx >= nvec) return;
  const float* op = o    + (size_t)idx * HD;
  const float* gp = gate + (size_t)idx * HD;
  float ss = 0.f;
  for (int i = 0; i < HD; ++i) { float v = op[i] * gp[i]; ss += v * v; }
  float r = rsqrtf(ss / (float)HD + 1e-6f);
  int col0 = (idx % H) * HD;
  bf16_t* out = ogb + (size_t)idx * HD;
  for (int i = 0; i < HD; ++i)
    out[i] = (bf16_t)(op[i] * gp[i] * r * gnorm[col0 + i]);
}

// ---------------------------------------------------------------------------
// Chunked delta-rule recurrence (WY form), chunk C=32, one WG per (b,h).
// S [128x128 f32] lives in LDS (~204KB dynamic LDS -> CDNA5's 320KB WGP LDS).
// Per chunk:
//   async-DMA the Q/K/V chunk into LDS (ASYNCcnt)
//   Pred = Kc @ S, QS = Qc @ S, Amat = Kc Kc^T, Bq = Qc Kc^T  (all WMMA)
//   forward-substitution (VALU, one thread per value-column, no intra-loop syncs)
//   S += Kc^T Wc  (WMMA, K=32 via transposed LDS tiles), refresh bf16 S^T mirror
// ---------------------------------------------------------------------------
#define D_SMEM 209024

__global__ void __launch_bounds__(256)
delta_kernel(const bf16_t* __restrict__ q, const bf16_t* __restrict__ k,
             const bf16_t* __restrict__ v, const float* __restrict__ beta,
             float* __restrict__ o, int B, int N, int H) {
  extern __shared__ char smem[];
  constexpr int HD = 128, C = 32;
  float*  S     = (float*)(smem);                 // [128][132] f32 master state
  bf16_t* Sbt   = (bf16_t*)(smem + 67584);        // [128][136] bf16, rows=j cols=i (S^T)
  bf16_t* Qc    = (bf16_t*)(smem + 102400);       // [32][136]
  bf16_t* Kc    = (bf16_t*)(smem + 111104);       // [32][136]
  bf16_t* Vc    = (bf16_t*)(smem + 119808);       // [32][136]
  bf16_t* KcT   = (bf16_t*)(smem + 128512);       // [128][40]
  bf16_t* WcT   = (bf16_t*)(smem + 138752);       // [128][40]
  float*  Pred  = (float*)(smem + 148992);        // [32][132]
  float*  Wc    = (float*)(smem + 165888);        // [32][132]
  float*  QS    = (float*)(smem + 182784);        // [32][132]
  float*  Amat  = (float*)(smem + 199680);        // [32][36]
  float*  Bq    = (float*)(smem + 204288);        // [32][36]
  float*  betac = (float*)(smem + 208896);        // [32]

  const int tid  = threadIdx.x;
  const int wave = tid >> 5;
  const int bh   = blockIdx.x;
  const int b    = bh / H, h = bh % H;
  const int D    = H * HD;

  for (int i = tid; i < 128 * 132; i += 256) S[i] = 0.f;
  for (int i = tid; i < 128 * 136; i += 256) Sbt[i] = (bf16_t)0.f;
  __syncthreads();

  const bf16_t* qb  = q + (size_t)b * N * D + h * HD;
  const bf16_t* kb  = k + (size_t)b * N * D + h * HD;
  const bf16_t* vb  = v + (size_t)b * N * D + h * HD;
  const float*  btp = beta + (size_t)b * N * H + h;
  float*        ob  = o + (size_t)b * N * D + h * HD;

  for (int c0 = 0; c0 < N; c0 += C) {
    // async-DMA chunk (32x128 each) straight into LDS, 16B granules
#pragma unroll
    for (int ii = 0; ii < 2; ++ii) {
      int i = tid + (ii << 8);
      int t = i >> 4, cc = (i & 15) << 3;
      uint32_t goff = (uint32_t)(((size_t)(c0 + t) * D + cc) * 2);
      async_b128(qb, goff, &Qc[t * 136 + cc]);
      async_b128(kb, goff, &Kc[t * 136 + cc]);
      async_b128(vb, goff, &Vc[t * 136 + cc]);
    }
    if (tid < C) betac[tid] = btp[(size_t)(c0 + tid) * H];
    wait_async0();
    __syncthreads();
    for (int i = tid; i < HD * C; i += 256) {  // K^T tile for the rank-32 update
      int ii = i >> 5, t = i & 31;
      KcT[ii * 40 + t] = Kc[t * 136 + ii];
    }
    __syncthreads();

    // Pred = Kc@S and QS = Qc@S : 16 tiles (2x8) each, 2 per wave, K=128 (4 steps)
    for (int tile = wave; tile < 16; tile += 8) {
      int mt = tile >> 3, nt = tile & 7;
      f32x8 ap = f32x8{0,0,0,0,0,0,0,0};
      f32x8 aq = f32x8{0,0,0,0,0,0,0,0};
#pragma unroll
      for (int kk = 0; kk < 4; ++kk) {
        bf16x16 bf = frag_ld(Sbt + kk * 32, 136, nt * 16);
        ap = wmma_bf16(frag_ld(Kc + kk * 32, 136, mt * 16), bf, ap);
        aq = wmma_bf16(frag_ld(Qc + kk * 32, 136, mt * 16), bf, aq);
      }
      cfrag_st(Pred, 132, mt * 16, nt * 16, ap);
      cfrag_st(QS,   132, mt * 16, nt * 16, aq);
    }
    // Amat = Kc Kc^T (waves 0-3), Bq = Qc Kc^T (waves 4-7): 2x2 tiles each
    {
      int t4 = wave & 3;
      int mt = t4 >> 1, nt = t4 & 1;
      const bf16_t* Ab = (wave < 4) ? Kc : Qc;
      float* Cd        = (wave < 4) ? Amat : Bq;
      f32x8 acc = f32x8{0,0,0,0,0,0,0,0};
#pragma unroll
      for (int kk = 0; kk < 4; ++kk)
        acc = wmma_bf16(frag_ld(Ab + kk * 32, 136, mt * 16),
                        frag_ld(Kc + kk * 32, 136, nt * 16), acc);
      cfrag_st(Cd, 36, mt * 16, nt * 16, acc);
    }
    __syncthreads();

    // intra-chunk forward substitution: thread j owns value-column j
    if (tid < HD) {
      const int j = tid;
      for (int t = 0; t < C; ++t) {
        float w  = (float)Vc[t * 136 + j] - Pred[t * 132 + j];
        float ov = QS[t * 132 + j];
        for (int s = 0; s < t; ++s) {
          float ws = Wc[s * 132 + j];
          w  -= Amat[t * 36 + s] * ws;
          ov += Bq[t * 36 + s] * ws;
        }
        w *= betac[t];
        Wc[t * 132 + j]  = w;
        WcT[j * 40 + t]  = (bf16_t)w;
        ov += Bq[t * 36 + t] * w;         // diagonal (s == t)
        ob[(size_t)(c0 + t) * D + j] = ov;
      }
    }
    __syncthreads();

    // S += Kc^T @ Wc : 64 tiles (8x8), 8 per wave, single K=32 WMMA each
    for (int tile = wave; tile < 64; tile += 8) {
      int it = tile >> 3, jt = tile & 7;
      f32x8 acc = cfrag_ld(S, 132, it * 16, jt * 16);
      acc = wmma_bf16(frag_ld(KcT, 40, it * 16), frag_ld(WcT, 40, jt * 16), acc);
      cfrag_st(S, 132, it * 16, jt * 16, acc);
    }
    __syncthreads();
    // refresh bf16 S^T mirror for next chunk's B-fragments
    for (int i = tid; i < HD * HD; i += 256) {
      int ii = i >> 7, jj = i & 127;
      Sbt[jj * 136 + ii] = (bf16_t)S[ii * 132 + jj];
    }
    __syncthreads();
  }
}

// ---------------------------------------------------------------------------
// Host launcher
// ---------------------------------------------------------------------------
extern "C" void kernel_launch(void* const* d_in, const int* in_sizes, int n_in,
                              void* d_out, int out_size, void* d_ws, size_t ws_size,
                              hipStream_t stream) {
  (void)in_sizes; (void)n_in; (void)out_size; (void)ws_size;
  const int Bb = 2, Nn = 2048, Dd = 2048, Hh = 16, HD = 128;
  const int Mm = Bb * Nn;  // 4096

  const float* x     = (const float*)d_in[0];
  const float* Wq    = (const float*)d_in[1];
  const float* Wk    = (const float*)d_in[2];
  const float* Wv    = (const float*)d_in[3];
  const float* Wbeta = (const float*)d_in[4];
  const float* Wg1   = (const float*)d_in[5];
  const float* Wg2   = (const float*)d_in[6];
  const float* gnorm = (const float*)d_in[7];
  const float* Wo    = (const float*)d_in[8];

  char* ws = (char*)d_ws;
  size_t off = 0;
  auto alloc = [&](size_t bytes) { size_t o = off; off += (bytes + 255) & ~(size_t)255; return o; };

  bf16_t* xb   = (bf16_t*)(ws + alloc((size_t)Mm * Dd * 2));
  bf16_t* Wqb  = (bf16_t*)(ws + alloc((size_t)Dd * Dd * 2));
  bf16_t* Wkb  = (bf16_t*)(ws + alloc((size_t)Dd * Dd * 2));
  bf16_t* Wvb  = (bf16_t*)(ws + alloc((size_t)Dd * Dd * 2));
  bf16_t* Wob  = (bf16_t*)(ws + alloc((size_t)Dd * Dd * 2));
  bf16_t* Wg1b = (bf16_t*)(ws + alloc((size_t)HD * Dd * 2));
  bf16_t* Wg2b = (bf16_t*)(ws + alloc((size_t)Dd * HD * 2));
  bf16_t* qbf  = (bf16_t*)(ws + alloc((size_t)Mm * Dd * 2));
  float*  kf   = (float*) (ws + alloc((size_t)Mm * Dd * 4));
  bf16_t* kbf  = (bf16_t*)(ws + alloc((size_t)Mm * Dd * 2));
  bf16_t* vbf  = (bf16_t*)(ws + alloc((size_t)Mm * Dd * 2));
  bf16_t* g1b  = (bf16_t*)(ws + alloc((size_t)Mm * HD * 2));
  float*  gate = (float*) (ws + alloc((size_t)Mm * Dd * 4));
  float*  betb = (float*) (ws + alloc((size_t)Mm * Hh * 4));
  float*  obuf = (float*) (ws + alloc((size_t)Mm * Dd * 4));
  bf16_t* ogb  = (bf16_t*)(ws + alloc((size_t)Mm * Dd * 2));

  auto cvt = [&](const float* src, bf16_t* dst, size_t n) {
    int blocks = (int)((n + 255) / 256);
    if (blocks > 16384) blocks = 16384;
    f2bf_kernel<<<blocks, 256, 0, stream>>>(src, dst, n);
  };
  cvt(x,   xb,   (size_t)Mm * Dd);
  cvt(Wq,  Wqb,  (size_t)Dd * Dd);
  cvt(Wk,  Wkb,  (size_t)Dd * Dd);
  cvt(Wv,  Wvb,  (size_t)Dd * Dd);
  cvt(Wo,  Wob,  (size_t)Dd * Dd);
  cvt(Wg1, Wg1b, (size_t)HD * Dd);
  cvt(Wg2, Wg2b, (size_t)Dd * HD);

  dim3 gFull(Dd / 128, Mm / 128);   // 16 x 32
  dim3 gNarrow(1, Mm / 128);        // N = 128

  // q = silu(x Wq^T) -> bf16 ; k = silu(x Wk^T) -> f32 (needs L2 norm) ; v -> bf16
  gemm_wmma<1, 1><<<gFull, 256, 0, stream>>>(xb, Wqb, nullptr, qbf, Mm, Dd, Dd);
  gemm_wmma<1, 0><<<gFull, 256, 0, stream>>>(xb, Wkb, kf, nullptr, Mm, Dd, Dd);
  gemm_wmma<0, 1><<<gFull, 256, 0, stream>>>(xb, Wvb, nullptr, vbf, Mm, Dd, Dd);
  // gate bottleneck: G1 = x Wg1^T (bf16), gate = sigmoid(G1 Wg2^T) (f32)
  gemm_wmma<0, 1><<<gNarrow, 256, 0, stream>>>(xb, Wg1b, nullptr, g1b, Mm, HD, Dd);
  gemm_wmma<2, 0><<<gFull, 256, 0, stream>>>(g1b, Wg2b, gate, nullptr, Mm, Dd, HD);

  beta_kernel<<<(Mm * Hh + 255) / 256, 256, 0, stream>>>(x, Wbeta, betb, Mm, Dd, Hh);
  knorm_kernel<<<(Mm * Hh + 255) / 256, 256, 0, stream>>>(kf, kbf, Mm * Hh, HD);

  delta_kernel<<<Bb * Hh, 256, D_SMEM, stream>>>(qbf, kbf, vbf, betb, obuf, Bb, Nn, Hh);

  og_kernel<<<(Mm * Hh + 255) / 256, 256, 0, stream>>>(obuf, gate, gnorm, ogb,
                                                       Mm * Hh, HD, Hh);
  gemm_wmma<0, 0><<<gFull, 256, 0, stream>>>(ogb, Wob, (float*)d_out, nullptr, Mm, Dd, Dd);
}
